// MambaBackbone_8400956031379
// MI455X (gfx1250) — matrix operand order repaired and checked by reading, not compiled, not run on hardware
//
#include <hip/hip_runtime.h>
#include <hip/hip_bf16.h>
#include <math.h>

// ---------------- problem constants (from reference) ----------------
#define BB        4
#define LL        4096
#define IN_DIM    256
#define D_MODEL   512
#define D_INNER   1024
#define D_STATE   16
#define D_CONV    4
#define DT_RANK   32
#define BL        (BB * LL)          // 16384 token rows

typedef __attribute__((ext_vector_type(16))) _Float16 v16h;
typedef __attribute__((ext_vector_type(8)))  _Float16 v8h;
typedef __attribute__((ext_vector_type(8)))  float    v8f;

union Frag { v16h v; v8h h[2]; };

// ---------------- conversion / transpose kernels ----------------
__global__ void cvt_f32_to_f16(const float* __restrict__ in,
                               _Float16* __restrict__ out, int n) {
  int i = blockIdx.x * blockDim.x + threadIdx.x;
  if (i < n) out[i] = (_Float16)in[i];
}

// in: [K,N] f32 row-major  ->  out: [N,K] f16 row-major (transposed)
__global__ void transpose_cvt(const float* __restrict__ in,
                              _Float16* __restrict__ out, int K, int N) {
  int i = blockIdx.x * blockDim.x + threadIdx.x;
  if (i >= K * N) return;
  int k = i / N, n = i - k * N;
  out[(size_t)n * K + k] = (_Float16)in[i];
}

// x_dbl[:, :32] f32 -> f16 (dt_proj WMMA A-operand)
__global__ void extract_dtlow(const float* __restrict__ xdbl,
                              _Float16* __restrict__ dtlow) {
  int i = blockIdx.x * blockDim.x + threadIdx.x;   // BL * 32
  if (i >= BL * DT_RANK) return;
  int r = i >> 5, c = i & 31;
  dtlow[i] = (_Float16)xdbl[(size_t)r * 64 + c];
}

// ---------------- WMMA GEMM: C[M,N] = A[M,K] * Bt[N,K]^T ----------------
// Register-blocked: one wave computes a 32x64 output tile
// (2 A-frags x 4 B-frags -> 8 v_wmma per K-step, 8 f32 accumulators).
// MODE 0: +bias, store f16     (input_proj -> h16)
// MODE 1: store f32            (in_proj, x_proj)
// MODE 2: softplus(acc+bias), store f32 (dt_proj)
// 8 waves / 256-thread block (wave32). Requires M%32==0, N%64==0, K%32==0.
template <int MODE>
__global__ void wmma_gemm(const _Float16* __restrict__ A,
                          const _Float16* __restrict__ Bt,
                          float* __restrict__ Cf,
                          _Float16* __restrict__ Ch,
                          const float* __restrict__ bias,
                          int M, int N, int K) {
  const int lane  = threadIdx.x & 31;
  const int wave  = threadIdx.x >> 5;
  const int tilesN = N >> 6;                          // 64-wide tiles
  const long tile   = (long)blockIdx.x * 8 + wave;
  const long nTiles = (long)(M >> 5) * tilesN;        // 32-tall tiles
  if (tile >= nTiles) return;                         // wave-uniform exit
  const int tm = (int)(tile / tilesN);
  const int tn = (int)(tile % tilesN);
  const int m0 = tm << 5;                             // 32 rows
  const int n0 = tn << 6;                             // 64 cols

  const int hi    = lane >> 4;                        // lane half
  const int lo16  = lane & 15;
  const int kaSel = hi << 3;                          // A K-offset: 0 / 8
  const int kbSel = hi << 3;                          // B K-offset: 0 / 8

  // A fragment row pointers (2 M-subtiles), B fragment col pointers (4 N-subtiles)
  const _Float16* ap0 = A + (size_t)(m0 + lo16) * K + kaSel;
  const _Float16* ap1 = ap0 + (size_t)16 * K;
  const _Float16* bp0 = Bt + (size_t)(n0 + lo16) * K + kbSel;
  const _Float16* bp1 = bp0 + (size_t)16 * K;
  const _Float16* bp2 = bp0 + (size_t)32 * K;
  const _Float16* bp3 = bp0 + (size_t)48 * K;

  v8f acc00 = {}, acc01 = {}, acc02 = {}, acc03 = {};
  v8f acc10 = {}, acc11 = {}, acc12 = {}, acc13 = {};

  for (int k0 = 0; k0 < K; k0 += 32) {
    __builtin_prefetch(ap0 + 32, 0, 0);               // global_prefetch_b8
    __builtin_prefetch(ap1 + 32, 0, 0);
    __builtin_prefetch(bp0 + 32, 0, 0);
    __builtin_prefetch(bp2 + 32, 0, 0);
    Frag a0, a1, b0, b1, b2, b3;
    // A 16x32 f16 layout: lane-half -> K {0..7,16..23} / {8..15,24..31}
    a0.h[0] = *(const v8h*)(ap0);  a0.h[1] = *(const v8h*)(ap0 + 16);
    a1.h[0] = *(const v8h*)(ap1);  a1.h[1] = *(const v8h*)(ap1 + 16);
    // B 32x16 f16 layout: lane-half -> K {0..7,16..23} / {8..15,24..31}
    b0.h[0] = *(const v8h*)(bp0);  b0.h[1] = *(const v8h*)(bp0 + 16);
    b1.h[0] = *(const v8h*)(bp1);  b1.h[1] = *(const v8h*)(bp1 + 16);
    b2.h[0] = *(const v8h*)(bp2);  b2.h[1] = *(const v8h*)(bp2 + 16);
    b3.h[0] = *(const v8h*)(bp3);  b3.h[1] = *(const v8h*)(bp3 + 16);

    acc00 = __builtin_amdgcn_wmma_f32_16x16x32_f16(false, a0.v, false, b0.v, (short)0, acc00, false, false);
    acc01 = __builtin_amdgcn_wmma_f32_16x16x32_f16(false, a0.v, false, b1.v, (short)0, acc01, false, false);
    acc02 = __builtin_amdgcn_wmma_f32_16x16x32_f16(false, a0.v, false, b2.v, (short)0, acc02, false, false);
    acc03 = __builtin_amdgcn_wmma_f32_16x16x32_f16(false, a0.v, false, b3.v, (short)0, acc03, false, false);
    acc10 = __builtin_amdgcn_wmma_f32_16x16x32_f16(false, a1.v, false, b0.v, (short)0, acc10, false, false);
    acc11 = __builtin_amdgcn_wmma_f32_16x16x32_f16(false, a1.v, false, b1.v, (short)0, acc11, false, false);
    acc12 = __builtin_amdgcn_wmma_f32_16x16x32_f16(false, a1.v, false, b2.v, (short)0, acc12, false, false);
    acc13 = __builtin_amdgcn_wmma_f32_16x16x32_f16(false, a1.v, false, b3.v, (short)0, acc13, false, false);

    ap0 += 32; ap1 += 32; bp0 += 32; bp1 += 32; bp2 += 32; bp3 += 32;
  }

  // C/D layout: VGPR r -> row (+ 8*lane_half), col = lane&15 within subtile
  v8f accs[2][4] = {{acc00, acc01, acc02, acc03}, {acc10, acc11, acc12, acc13}};
  #pragma unroll
  for (int i = 0; i < 2; ++i) {
    const int row0 = m0 + (i << 4) + (hi << 3);
    #pragma unroll
    for (int j = 0; j < 4; ++j) {
      const int col = n0 + (j << 4) + lo16;
      const float bv = (MODE != 1 && bias) ? bias[col] : 0.0f;
      #pragma unroll
      for (int r = 0; r < 8; ++r) {
        float v = accs[i][j][r];
        const size_t idx = (size_t)(row0 + r) * N + col;
        if (MODE == 0) {
          Ch[idx] = (_Float16)(v + bv);
        } else if (MODE == 1) {
          Cf[idx] = v;
        } else {               // softplus(v + bias)
          v += bv;
          v = (v > 0.0f) ? (v + log1pf(expf(-v))) : log1pf(expf(v));
          Cf[idx] = v;
        }
      }
    }
  }
}

// ---------------- depthwise causal conv1d + SiLU ----------------
// u_raw = xz[:, :D_INNER];  u(b,l,e) = silu(sum_j cw[e,j]*u_raw(b,l-3+j,e)+cb[e])
__global__ void conv_silu(const float* __restrict__ xz,
                          const float* __restrict__ cw,
                          const float* __restrict__ cb,
                          float* __restrict__ u32o,
                          _Float16* __restrict__ u16o) {
  int i = blockIdx.x * blockDim.x + threadIdx.x;     // BL * D_INNER
  if (i >= BL * D_INNER) return;
  int e  = i & (D_INNER - 1);
  int bl = i >> 10;
  int l  = bl & (LL - 1);
  int b  = bl >> 12;
  float acc = cb[e];
  #pragma unroll
  for (int j = 0; j < D_CONV; ++j) {
    int ls = l - (D_CONV - 1) + j;
    if (ls >= 0)
      acc += cw[e * D_CONV + j] * xz[(size_t)(b * LL + ls) * (2 * D_INNER) + e];
  }
  float s = acc / (1.0f + expf(-acc));               // SiLU
  u32o[i] = s;
  u16o[i] = (_Float16)s;
}

// ---------------- selective scan (wave-parallel over channels) ----------------
// 16 lanes = one (b,e) channel, lane = state n. Sequential over L.
// Fuses D-skip, z-gate, and the mean over L (linear out_proj commutes with mean).
__global__ void selective_scan(const float* __restrict__ dtbuf,   // [BL,E]
                               const float* __restrict__ ubuf,    // [BL,E]
                               const float* __restrict__ xdbl,    // [BL,64]
                               const float* __restrict__ xz,      // [BL,2E]
                               const float* __restrict__ A_log,   // [E,N]
                               const float* __restrict__ D_skip,  // [E]
                               float* __restrict__ ybar) {        // [B,E]
  int gid = blockIdx.x * blockDim.x + threadIdx.x;   // BB*E*N = 65536 lanes
  int n  = gid & (D_STATE - 1);
  int ch = gid >> 4;                                 // 0..4095
  int e  = ch & (D_INNER - 1);
  int b  = ch >> 10;
  const float Aen = -expf(A_log[e * D_STATE + n]);
  const float Dv  = D_skip[e];
  float h = 0.0f, ysum = 0.0f;
  const size_t base = (size_t)b * LL;
  for (int t = 0; t < LL; ++t) {
    const size_t row = base + t;
    const float dt = dtbuf[row * D_INNER + e];       // broadcast in 16-lane group
    const float uu = ubuf[row * D_INNER + e];
    const float Bn = xdbl[row * 64 + DT_RANK + n];
    const float Cn = xdbl[row * 64 + DT_RANK + D_STATE + n];
    h = expf(dt * Aen) * h + dt * uu * Bn;
    float y = h * Cn;
    y += __shfl_xor(y, 1, 32);                       // reduce over 16 states
    y += __shfl_xor(y, 2, 32);
    y += __shfl_xor(y, 4, 32);
    y += __shfl_xor(y, 8, 32);
    if (n == 0) {
      const float z = xz[row * (2 * D_INNER) + D_INNER + e];
      const float g = z / (1.0f + expf(-z));         // silu(z)
      ysum += (y + uu * Dv) * g;
    }
  }
  if (n == 0) ybar[b * D_INNER + e] = ysum * (1.0f / LL);
}

// ---------------- final tiny projection: out = ybar @ out_w ----------------
__global__ void out_proj(const float* __restrict__ ybar,
                         const float* __restrict__ out_w,
                         float* __restrict__ out) {
  int i = blockIdx.x * blockDim.x + threadIdx.x;     // BB * D_MODEL
  if (i >= BB * D_MODEL) return;
  int m = i & (D_MODEL - 1);
  int b = i >> 9;
  float acc = 0.0f;
  for (int e = 0; e < D_INNER; ++e)
    acc += ybar[b * D_INNER + e] * out_w[(size_t)e * D_MODEL + m];
  out[i] = acc;
}

// ---------------- host-side launch ----------------
extern "C" void kernel_launch(void* const* d_in, const int* in_sizes, int n_in,
                              void* d_out, int out_size, void* d_ws, size_t ws_size,
                              hipStream_t stream) {
  const float* x       = (const float*)d_in[0];   // [B,L,IN_DIM]
  const float* w_proj  = (const float*)d_in[1];   // [IN_DIM,D_MODEL]
  const float* b_proj  = (const float*)d_in[2];   // [D_MODEL]
  const float* in_w    = (const float*)d_in[3];   // [D_MODEL,2*D_INNER]
  const float* conv_w  = (const float*)d_in[4];   // [D_INNER,D_CONV]
  const float* conv_b  = (const float*)d_in[5];   // [D_INNER]
  const float* xproj_w = (const float*)d_in[6];   // [D_INNER,64]
  const float* dt_w    = (const float*)d_in[7];   // [DT_RANK,D_INNER]
  const float* dt_b    = (const float*)d_in[8];   // [D_INNER]
  const float* A_log   = (const float*)d_in[9];   // [D_INNER,D_STATE]
  const float* D_skip  = (const float*)d_in[10];  // [D_INNER]
  const float* out_w   = (const float*)d_in[11];  // [D_INNER,D_MODEL]
  float* out = (float*)d_out;                     // [B,D_MODEL]

  // ---- workspace layout (all offsets multiples of 256 bytes) ----
  char* ws = (char*)d_ws;
  size_t off = 0;
  auto alloc = [&](size_t bytes) { char* p = ws + off; off += (bytes + 255) & ~size_t(255); return p; };
  _Float16* xh     = (_Float16*)alloc((size_t)BL * IN_DIM * 2);        // 8 MB
  _Float16* wpT    = (_Float16*)alloc((size_t)D_MODEL * IN_DIM * 2);   // 256 KB
  _Float16* h16    = (_Float16*)alloc((size_t)BL * D_MODEL * 2);       // 16 MB
  _Float16* inwT   = (_Float16*)alloc((size_t)2 * D_INNER * D_MODEL * 2); // 2 MB
  float*    xz     = (float*)   alloc((size_t)BL * 2 * D_INNER * 4);   // 128 MB
  float*    u32    = (float*)   alloc((size_t)BL * D_INNER * 4);       // 64 MB
  _Float16* u16    = (_Float16*)alloc((size_t)BL * D_INNER * 2);       // 32 MB
  _Float16* xpT    = (_Float16*)alloc((size_t)64 * D_INNER * 2);       // 128 KB
  float*    xdbl   = (float*)   alloc((size_t)BL * 64 * 4);            // 4 MB
  _Float16* dtlow  = (_Float16*)alloc((size_t)BL * DT_RANK * 2);       // 1 MB
  _Float16* dtwT   = (_Float16*)alloc((size_t)D_INNER * DT_RANK * 2);  // 64 KB
  float*    dtbuf  = (float*)   alloc((size_t)BL * D_INNER * 4);       // 64 MB
  float*    ybar   = (float*)   alloc((size_t)BB * D_INNER * 4);       // 16 KB
  (void)ws_size; (void)in_sizes; (void)n_in; (void)out_size;

  const int T = 256;
  auto nb = [](long n, int t) { return (int)((n + t - 1) / t); };
  auto gemm_blocks = [](int M, int N) { long tiles = (long)(M >> 5) * (N >> 6);
                                        return (int)((tiles + 7) / 8); };

  // 1) f32->f16 conversions / weight transposes
  cvt_f32_to_f16<<<nb((long)BL * IN_DIM, T), T, 0, stream>>>(x, xh, BL * IN_DIM);
  transpose_cvt<<<nb((long)IN_DIM * D_MODEL, T), T, 0, stream>>>(w_proj, wpT, IN_DIM, D_MODEL);
  transpose_cvt<<<nb((long)D_MODEL * 2 * D_INNER, T), T, 0, stream>>>(in_w, inwT, D_MODEL, 2 * D_INNER);
  transpose_cvt<<<nb((long)D_INNER * 64, T), T, 0, stream>>>(xproj_w, xpT, D_INNER, 64);
  transpose_cvt<<<nb((long)DT_RANK * D_INNER, T), T, 0, stream>>>(dt_w, dtwT, DT_RANK, D_INNER);

  // 2) input_proj: h16 = f16(x @ w_proj + b_proj)        [16384,512]
  wmma_gemm<0><<<gemm_blocks(BL, D_MODEL), T, 0, stream>>>(
      xh, wpT, nullptr, h16, b_proj, BL, D_MODEL, IN_DIM);

  // 3) in_proj: xz = h @ in_w                            [16384,2048]
  wmma_gemm<1><<<gemm_blocks(BL, 2 * D_INNER), T, 0, stream>>>(
      h16, inwT, xz, nullptr, nullptr, BL, 2 * D_INNER, D_MODEL);

  // 4) depthwise causal conv + SiLU -> u (f32 + f16)
  conv_silu<<<nb((long)BL * D_INNER, T), T, 0, stream>>>(xz, conv_w, conv_b, u32, u16);

  // 5) x_proj: x_dbl = u @ xproj_w                       [16384,64]
  wmma_gemm<1><<<gemm_blocks(BL, 64), T, 0, stream>>>(
      u16, xpT, xdbl, nullptr, nullptr, BL, 64, D_INNER);

  // 6) dt_proj: dt = softplus(x_dbl[:,:32] @ dt_w + dt_b) [16384,1024]
  extract_dtlow<<<nb((long)BL * DT_RANK, T), T, 0, stream>>>(xdbl, dtlow);
  wmma_gemm<2><<<gemm_blocks(BL, D_INNER), T, 0, stream>>>(
      dtlow, dtwT, dtbuf, nullptr, dt_b, BL, D_INNER, DT_RANK);

  // 7) selective scan (fused D-skip, z-gate, mean over L) -> ybar [4,1024]
  selective_scan<<<nb((long)BB * D_INNER * D_STATE, T), T, 0, stream>>>(
      dtbuf, u32, xdbl, xz, A_log, D_skip, ybar);

  // 8) out = ybar @ out_w                                 [4,512]
  out_proj<<<nb((long)BB * D_MODEL, T), T, 0, stream>>>(ybar, out_w, out);
}
